// AttentionRNNCell_21526376087583
// MI455X (gfx1250) — matrix operand order repaired and checked by reading, not compiled, not run on hardware
//
#include <hip/hip_runtime.h>

// ---------------------------------------------------------------------------
// AttentionRNNCell fused pipeline for gfx1250 (MI455X), bf16 WMMA GEMMs.
// softmax over k_len==1 is identically 1 => attn == kx; w_q/b_q/mlp_w dead.
//
//   kx        = inputs  @ w_k^T    + b_k          [B,1024]
//   attn_out  = kx      @ w_proj^T + b_proj       [B,1024]  -> ffn_in[:,1024:]
//   h1        = relu(ffn_in @ w1^T + b1)          [B,4096]
//   ffn_out   = h1      @ w2^T     + b2           [B,2048]  -> d_out[0:...]
//   new_hid   = combined@ w_out^T  + b_out        [B,1024]  -> d_out[B*2048:]
//   d_out[0:B*2048] = log_softmax(ffn_out) rows of 2048 (in place)
// ---------------------------------------------------------------------------

typedef __attribute__((ext_vector_type(16))) __bf16 v16bf;
typedef __attribute__((ext_vector_type(8)))  float  v8f;
typedef __attribute__((ext_vector_type(4)))  unsigned int v4u;
typedef __attribute__((ext_vector_type(4)))  int v4i;

#define BM 128
#define BN 256
#define BK 32
#define LDS_STRIDE 40   // 32 + 8 bf16 pad -> 80B row pitch, 16B aligned

#define AS_GLOBAL __attribute__((address_space(1)))
#define AS_LDS    __attribute__((address_space(3)))

#if defined(__AMDGCN__) && __has_builtin(__builtin_amdgcn_global_load_async_to_lds_b128)
#define HAVE_ASYNC 1
#else
#define HAVE_ASYNC 0
#endif

// ---------------------------------------------------------------------------
// Tiled bf16 GEMM:  out[M,N] = act(A[M,K](lda) * W[N,K](ldw)^T + bias[N])
// 256 threads = 8 waves as 2(M) x 4(N); per-wave 64x64 = 4x4 WMMA tiles.
// Double-buffered LDS, one barrier per K-step; async-to-LDS staging if the
// toolchain exposes the gfx1250 async builtins, else load+ds_store fallback.
// ---------------------------------------------------------------------------
template <bool RELU, bool OUT_BF16>
__global__ __launch_bounds__(256) void gemm_bf16_wmma(
    const __bf16* __restrict__ A, int lda,
    const __bf16* __restrict__ W, int ldw,
    const float* __restrict__ bias,
    float* __restrict__ outF, __bf16* __restrict__ outB, int ldc,
    int M, int N, int K)
{
  __shared__ __align__(16) __bf16 sA[2][BM * LDS_STRIDE];
  __shared__ __align__(16) __bf16 sB[2][BN * LDS_STRIDE];

  const int t      = threadIdx.x;
  const int lane   = t & 31;
  const int wave   = t >> 5;        // 0..7
  const int waveM  = wave >> 2;     // 0..1  (2 waves along M)
  const int waveN  = wave & 3;      // 0..3  (4 waves along N)
  const int blockM = blockIdx.y * BM;
  const int blockN = blockIdx.x * BN;
  const int wm0    = waveM * 64;
  const int wn0    = waveN * 64;
  const int lh     = lane & 15;
  const int lg     = lane >> 4;

  // staging map: 16B chunk per (row, 8-col group); thread t covers rows
  // sRow + p*64 (A: p<2, B: p<4), cols (t&3)*8 .. +7
  const int sRow = t >> 2;          // 0..63
  const int sCol = (t & 3) * 8;

  const __bf16* aSrc = A + (size_t)(blockM + sRow) * lda + sCol;
  const __bf16* bSrc = W + (size_t)(blockN + sRow) * ldw + sCol;

  v8f acc[4][4] = {};

#if !HAVE_ASYNC
  v4u ra[2], rb[4];
#endif

  auto issue = [&](int kt) {
    const int k0 = kt * BK;
#if HAVE_ASYNC
    const int buf = kt & 1;
#pragma unroll
    for (int p = 0; p < 2; ++p)
      __builtin_amdgcn_global_load_async_to_lds_b128(
          (AS_GLOBAL v4i*)(aSrc + (size_t)p * 64 * lda + k0),
          (AS_LDS v4i*)&sA[buf][(sRow + p * 64) * LDS_STRIDE + sCol], 0, 0);
#pragma unroll
    for (int p = 0; p < 4; ++p)
      __builtin_amdgcn_global_load_async_to_lds_b128(
          (AS_GLOBAL v4i*)(bSrc + (size_t)p * 64 * ldw + k0),
          (AS_LDS v4i*)&sB[buf][(sRow + p * 64) * LDS_STRIDE + sCol], 0, 0);
#else
#pragma unroll
    for (int p = 0; p < 2; ++p)
      ra[p] = *(const v4u*)(aSrc + (size_t)p * 64 * lda + k0);
#pragma unroll
    for (int p = 0; p < 4; ++p)
      rb[p] = *(const v4u*)(bSrc + (size_t)p * 64 * ldw + k0);
#endif
  };

  auto commit = [&](int kt) {
    const int buf = kt & 1; (void)buf;
#if HAVE_ASYNC
#if __has_builtin(__builtin_amdgcn_s_wait_asynccnt)
    __builtin_amdgcn_s_wait_asynccnt(0);
#else
    asm volatile("s_wait_asynccnt 0" ::: "memory");
#endif
#else
#pragma unroll
    for (int p = 0; p < 2; ++p)
      *(v4u*)&sA[buf][(sRow + p * 64) * LDS_STRIDE + sCol] = ra[p];
#pragma unroll
    for (int p = 0; p < 4; ++p)
      *(v4u*)&sB[buf][(sRow + p * 64) * LDS_STRIDE + sCol] = rb[p];
#endif
  };

  const int nk = K / BK;
  issue(0);

  for (int kt = 0; kt < nk; ++kt) {
    const int buf = kt & 1;
    commit(kt);           // async: own group complete; sync: ds_store regs
    __syncthreads();      // all waves' K-slice kt resident; prev reads done
    if (kt + 1 < nk) issue(kt + 1);   // prefetch into other buffer

    // Operand fragments per ISA 7.12.2 (bf16, K-contiguous per lane):
    //  A 16x32: lane<16 -> M=lane, K = {lg*8..+7} and {lg*8+16..+7}
    //  B 32x16: N=lane%16, K = {lg*8..+7} and {lg*8+16..+7}
    union Frag { v4u q[2]; v16bf v; };
    Frag bfr[4];
#pragma unroll
    for (int ni = 0; ni < 4; ++ni) {
      const __bf16* base = &sB[buf][(wn0 + ni * 16 + lh) * LDS_STRIDE + lg * 8];
      bfr[ni].q[0] = *(const v4u*)base;
      bfr[ni].q[1] = *(const v4u*)(base + 16);
    }
#pragma unroll
    for (int mi = 0; mi < 4; ++mi) {
      Frag af;
      const __bf16* base = &sA[buf][(wm0 + mi * 16 + lh) * LDS_STRIDE + lg * 8];
      af.q[0] = *(const v4u*)base;
      af.q[1] = *(const v4u*)(base + 16);
#pragma unroll
      for (int ni = 0; ni < 4; ++ni)
        acc[mi][ni] = __builtin_amdgcn_wmma_f32_16x16x32_bf16(
            false, af.v, false, bfr[ni].v, (short)0, acc[mi][ni],
            false, false);
    }
  }

  // Epilogue: C/D layout — lane holds col n=lane%16; VGPR j -> row j+8*(lane/16)
#pragma unroll
  for (int ni = 0; ni < 4; ++ni) {
    const int col = blockN + wn0 + ni * 16 + lh;
    const float bv = bias[col];
#pragma unroll
    for (int mi = 0; mi < 4; ++mi) {
#pragma unroll
      for (int j = 0; j < 8; ++j) {
        const int row = blockM + wm0 + mi * 16 + lg * 8 + j;
        float v = acc[mi][ni][j] + bv;
        if (RELU) v = v > 0.f ? v : 0.f;
        if (OUT_BF16) outB[(size_t)row * ldc + col] = (__bf16)v;
        else          outF[(size_t)row * ldc + col] = v;
      }
    }
  }
}

// ---------------------------------------------------------------------------
// f32 -> bf16 converters
// ---------------------------------------------------------------------------
__global__ __launch_bounds__(256) void cvt_f32_bf16(
    const float* __restrict__ src, __bf16* __restrict__ dst, size_t n)
{
  size_t i = (size_t)blockIdx.x * 256 + threadIdx.x;
  const size_t stride = (size_t)gridDim.x * 256;
  for (; i < n; i += stride) dst[i] = (__bf16)src[i];
}

__global__ __launch_bounds__(256) void cvt2d_f32_bf16(
    const float* __restrict__ src, int sld,
    __bf16* __restrict__ dst, int dld, int cols)
{
  const int r = blockIdx.y;
  for (int c = blockIdx.x * 256 + threadIdx.x; c < cols; c += gridDim.x * 256)
    dst[(size_t)r * dld + c] = (__bf16)src[(size_t)r * sld + c];
}

// ---------------------------------------------------------------------------
// in-place row-wise log_softmax, one 256-thread block per row
// ---------------------------------------------------------------------------
__global__ __launch_bounds__(256) void log_softmax_rows(
    float* __restrict__ data, int ncols)
{
  __shared__ float red[256];
  float* row = data + (size_t)blockIdx.x * ncols;
  const int t = threadIdx.x;

  float m = -3.4e38f;
  for (int c = t; c < ncols; c += 256) m = fmaxf(m, row[c]);
  red[t] = m; __syncthreads();
  for (int s = 128; s > 0; s >>= 1) {
    if (t < s) red[t] = fmaxf(red[t], red[t + s]);
    __syncthreads();
  }
  m = red[0]; __syncthreads();

  float sum = 0.f;
  for (int c = t; c < ncols; c += 256) sum += __expf(row[c] - m);
  red[t] = sum; __syncthreads();
  for (int s = 128; s > 0; s >>= 1) {
    if (t < s) red[t] += red[t + s];
    __syncthreads();
  }
  const float lse = m + __logf(red[0]);
  __syncthreads();

  for (int c = t; c < ncols; c += 256) row[c] -= lse;
}

// ---------------------------------------------------------------------------
extern "C" void kernel_launch(void* const* d_in, const int* in_sizes, int n_in,
                              void* d_out, int out_size, void* d_ws, size_t ws_size,
                              hipStream_t stream) {
  (void)in_sizes; (void)n_in; (void)out_size; (void)ws_size;

  constexpr int B = 4096, D = 1024, H = 1024, Adim = 1024, F = 4096;
  constexpr int DA = D + Adim;   // 2048
  constexpr int DH = D + H;      // 2048

  const float* inputs = (const float*)d_in[0];
  const float* hidden = (const float*)d_in[1];
  const float* w_k    = (const float*)d_in[2];
  const float* b_k    = (const float*)d_in[3];
  // d_in[4..6] (w_q, b_q, mlp_w): dead — softmax over k_len==1 is identically 1
  const float* w_proj = (const float*)d_in[7];
  const float* b_proj = (const float*)d_in[8];
  const float* ffn_w1 = (const float*)d_in[9];
  const float* ffn_b1 = (const float*)d_in[10];
  const float* ffn_w2 = (const float*)d_in[11];
  const float* ffn_b2 = (const float*)d_in[12];
  const float* w_out  = (const float*)d_in[13];
  const float* b_out  = (const float*)d_in[14];

  float* out_ls = (float*)d_out;                       // [B, DA] log-softmax
  float* out_nh = (float*)d_out + (size_t)B * DA;      // [B, H ] new hidden

  // workspace carve-up (bf16 buffers), all offsets 256B aligned
  char* ws = (char*)d_ws;
  __bf16* ffn_in   = (__bf16*)(ws);                               // [B, DA]
  __bf16* combined = (__bf16*)(ws + 16777216);                    // [B, DH]
  __bf16* kx       = (__bf16*)(ws + 33554432);                    // [B, H]
  __bf16* h1       = (__bf16*)(ws + 41943040);                    // [B, F]
  __bf16* wk_b     = (__bf16*)(ws + 75497472);                    // [H, D]
  __bf16* wproj_b  = (__bf16*)(ws + 77594624);                    // [A, H]
  __bf16* w1_b     = (__bf16*)(ws + 79691776);                    // [F, DA]
  __bf16* w2_b     = (__bf16*)(ws + 96468992);                    // [DA, F]
  __bf16* wout_b   = (__bf16*)(ws + 113246208);                   // [H, DH]

  // --- precision down-convert (once per call; deterministic) ---
  cvt_f32_bf16<<<2048, 256, 0, stream>>>(w_k,    wk_b,    (size_t)H * D);
  cvt_f32_bf16<<<2048, 256, 0, stream>>>(w_proj, wproj_b, (size_t)Adim * H);
  cvt_f32_bf16<<<4096, 256, 0, stream>>>(ffn_w1, w1_b,    (size_t)F * DA);
  cvt_f32_bf16<<<4096, 256, 0, stream>>>(ffn_w2, w2_b,    (size_t)DA * F);
  cvt_f32_bf16<<<2048, 256, 0, stream>>>(w_out,  wout_b,  (size_t)H * DH);

  // concats materialized by strided conversion into column slices
  cvt2d_f32_bf16<<<dim3(4, B), 256, 0, stream>>>(inputs, D, ffn_in,       DA, D);
  cvt2d_f32_bf16<<<dim3(4, B), 256, 0, stream>>>(inputs, D, combined,     DH, D);
  cvt2d_f32_bf16<<<dim3(4, B), 256, 0, stream>>>(hidden, H, combined + D, DH, H);

  // --- GEMM chain (block tile 128x256) ---
  // kx = inputs @ w_k^T + b_k   (A = ffn_in[:, :D] via lda=DA)
  gemm_bf16_wmma<false, true><<<dim3(H / BN, B / BM), 256, 0, stream>>>(
      ffn_in, DA, wk_b, D, b_k, nullptr, kx, H, B, H, D);

  // attn_out = kx @ w_proj^T + b_proj  -> ffn_in[:, D:] (ldc=DA)
  gemm_bf16_wmma<false, true><<<dim3(Adim / BN, B / BM), 256, 0, stream>>>(
      kx, H, wproj_b, H, b_proj, nullptr, ffn_in + D, DA, B, Adim, H);

  // h1 = relu(ffn_in @ w1^T + b1)
  gemm_bf16_wmma<true, true><<<dim3(F / BN, B / BM), 256, 0, stream>>>(
      ffn_in, DA, w1_b, DA, ffn_b1, nullptr, h1, F, B, F, DA);

  // ffn_out = h1 @ w2^T + b2  -> d_out (f32)
  gemm_bf16_wmma<false, false><<<dim3(DA / BN, B / BM), 256, 0, stream>>>(
      h1, F, w2_b, F, ffn_b2, out_ls, nullptr, DA, B, DA, F);

  // new_hidden = combined @ w_out^T + b_out -> d_out tail (f32)
  gemm_bf16_wmma<false, false><<<dim3(H / BN, B / BM), 256, 0, stream>>>(
      combined, DH, wout_b, DH, b_out, out_nh, nullptr, H, B, H, DH);

  // log_softmax over rows of ffn_out, in place
  log_softmax_rows<<<B, 256, 0, stream>>>(out_ls, DA);
}
